// TransformerDecoder_14611478741142
// MI455X (gfx1250) — matrix operand order repaired and checked by reading, not compiled, not run on hardware
//
#include <hip/hip_runtime.h>

// ---------------------------------------------------------------------------
// CDNA5 (gfx1250) transformer decoder forward.  All matmuls run on
// v_wmma_f32_16x16x32_bf16 (fp32 -> bf16 inputs, fp32 accumulate).
// ---------------------------------------------------------------------------

typedef __bf16 bf16;
typedef __attribute__((ext_vector_type(16))) __bf16 v16bf;
typedef __attribute__((ext_vector_type(8)))  __bf16 v8bf;
typedef __attribute__((ext_vector_type(8)))  float  v8f;

#define D_MODEL 1024
#define FFN_H   4096
#define N_LAYER 8
#define NQ_H    16
#define NKV_H   4
#define HEAD_D  64
#define BATCH   2
#define SEQ     1024
#define M_ROWS  (BATCH * SEQ)   // 2048 token rows

__device__ __forceinline__ v8f zero8() {
  v8f z;
#pragma unroll
  for (int i = 0; i < 8; ++i) z[i] = 0.f;
  return z;
}

__device__ __forceinline__ v16bf pack16(v8bf lo, v8bf hi) {
  v16bf r;
#pragma unroll
  for (int i = 0; i < 8; ++i) { r[i] = lo[i]; r[i + 8] = hi[i]; }
  return r;
}

// ---------------------------------------------------------------------------
// WMMA bf16 GEMM:  C[M,N] = A[M,K] @ W[K,N] + bias (+ residual)
// block = 256 threads (8 waves), tile 128x128, K-step 32.
// Wave w owns a 32x64 sub-tile = 2x4 WMMA 16x16 tiles.
// Register double-buffered staging: next tile's global loads overlap the
// current tile's LDS fragment loads + 8 WMMAs.
// ---------------------------------------------------------------------------
__global__ void __launch_bounds__(256) gemm_bf16(
    const float* __restrict__ A, const float* __restrict__ W,
    const float* __restrict__ bias, const float* __restrict__ res,
    float* __restrict__ C, int M, int N, int K)
{
  __shared__ bf16 As[128][40];   // [m][k], padded rows (80B, 16B aligned)
  __shared__ bf16 Bs[128][40];   // [n][k] (transposed so B-frags are contiguous)
  const int tid  = threadIdx.x;
  const int wave = tid >> 5;
  const int lane = tid & 31;
  const int lh   = lane >> 4;    // half-wave select
  const int l16  = lane & 15;
  const int m0   = blockIdx.y * 128;
  const int n0   = blockIdx.x * 128;
  const int wrow = (wave >> 1) * 32;   // 4 wave-rows
  const int wcol = (wave & 1) * 64;    // 2 wave-cols

  // staging thread mapping
  const int ar  = tid >> 3;            // A row 0..31 (+i*32)
  const int ac  = (tid & 7) * 4;       // A col 0..28
  const int bkr = tid >> 5;            // W k-row 0..7 (+i*8)
  const int bnc = (tid & 31) * 4;      // W n-col 0..124

  v8f acc[2][4];
#pragma unroll
  for (int mt = 0; mt < 2; ++mt)
#pragma unroll
    for (int nt = 0; nt < 4; ++nt) acc[mt][nt] = zero8();

  // prologue: load K-tile 0 into registers
  float4 ra[4], rb[4];
#pragma unroll
  for (int i = 0; i < 4; ++i) {
    ra[i] = *(const float4*)(A + (size_t)(m0 + ar + i * 32) * K + ac);
    rb[i] = *(const float4*)(W + (size_t)(bkr + i * 8) * N + n0 + bnc);
  }

  for (int kb = 0; kb < K; kb += 32) {
    // commit staged registers to LDS (fp32 -> bf16)
#pragma unroll
    for (int i = 0; i < 4; ++i) {
      const int r = ar + i * 32;
      As[r][ac + 0] = (bf16)ra[i].x; As[r][ac + 1] = (bf16)ra[i].y;
      As[r][ac + 2] = (bf16)ra[i].z; As[r][ac + 3] = (bf16)ra[i].w;
      const int kr = bkr + i * 8;
      Bs[bnc + 0][kr] = (bf16)rb[i].x; Bs[bnc + 1][kr] = (bf16)rb[i].y;
      Bs[bnc + 2][kr] = (bf16)rb[i].z; Bs[bnc + 3][kr] = (bf16)rb[i].w;
    }
    __syncthreads();

    // issue next K-tile's global loads (overlaps the WMMAs below)
    const int kn = kb + 32;
    if (kn < K) {
#pragma unroll
      for (int i = 0; i < 4; ++i) {
        ra[i] = *(const float4*)(A + (size_t)(m0 + ar + i * 32) * K + kn + ac);
        rb[i] = *(const float4*)(W + (size_t)(kn + bkr + i * 8) * N + n0 + bnc);
      }
    }
    // prefetch the tile after that into GL2 (global_prefetch_b8)
    if (kn + 32 < K) {
      __builtin_prefetch(A + (size_t)(m0 + ar) * K + kn + 32 + ac, 0, 1);
      __builtin_prefetch(W + (size_t)(kn + 32 + bkr) * N + n0 + bnc, 0, 1);
    }

    // A fragments (ISA 16-bit A layout: e<8 -> K=lh*8+e, e>=8 -> K=16+lh*8+e-8)
    v16bf af[2];
#pragma unroll
    for (int mt = 0; mt < 2; ++mt) {
      const bf16* p = &As[wrow + mt * 16 + l16][lh * 8];
      v8bf lo = *(const v8bf*)p;
      v8bf hi = *(const v8bf*)(p + 16);
      af[mt] = pack16(lo, hi);
    }
    // B fragments (same K order as A: e<8 -> K=lh*8+e, e>=8 -> K=16+lh*8+e-8; N = l16)
    v16bf bfrag[4];
#pragma unroll
    for (int nt = 0; nt < 4; ++nt) {
      const bf16* p = &Bs[wcol + nt * 16 + l16][lh * 8];
      v8bf lo = *(const v8bf*)p;
      v8bf hi = *(const v8bf*)(p + 16);
      bfrag[nt] = pack16(lo, hi);
    }
#pragma unroll
    for (int mt = 0; mt < 2; ++mt)
#pragma unroll
      for (int nt = 0; nt < 4; ++nt)
        acc[mt][nt] = __builtin_amdgcn_wmma_f32_16x16x32_bf16(
            false, af[mt], false, bfrag[nt], (short)0, acc[mt][nt], false, false);
    __syncthreads();
  }

  // epilogue: bias (+ residual).  C/D layout: lane holds N=l16, M=r+8*lh.
  // Residual check is uniform -> hoist it out of the element loops.
  float bb[4];
#pragma unroll
  for (int nt = 0; nt < 4; ++nt) bb[nt] = bias[n0 + wcol + nt * 16 + l16];

  if (res != nullptr) {
#pragma unroll
    for (int mt = 0; mt < 2; ++mt)
#pragma unroll
      for (int r = 0; r < 8; ++r) {
        const size_t rowb = (size_t)(m0 + wrow + mt * 16 + r + 8 * lh) * N
                          + n0 + wcol + l16;
#pragma unroll
        for (int nt = 0; nt < 4; ++nt)
          C[rowb + nt * 16] = acc[mt][nt][r] + bb[nt] + res[rowb + nt * 16];
      }
  } else {
#pragma unroll
    for (int mt = 0; mt < 2; ++mt)
#pragma unroll
      for (int r = 0; r < 8; ++r) {
        const size_t rowb = (size_t)(m0 + wrow + mt * 16 + r + 8 * lh) * N
                          + n0 + wcol + l16;
#pragma unroll
        for (int nt = 0; nt < 4; ++nt)
          C[rowb + nt * 16] = acc[mt][nt][r] + bb[nt];
      }
  }
}

// ---------------------------------------------------------------------------
// Flash attention, one wave per 16-query tile, 32-key chunks, causal.
// q,k: bf16 [B,S,nh,HD];  vT: bf16 [B,NKV,HD,S];  out: fp32 [B,S,NQ,HD]
// ---------------------------------------------------------------------------
__global__ void __launch_bounds__(256) attn_kernel(
    const bf16* __restrict__ q, const bf16* __restrict__ k,
    const bf16* __restrict__ vT, float* __restrict__ out)
{
  __shared__ bf16 Plds[8][16][40];     // per-wave P re-layout scratch
  const int tid  = threadIdx.x;
  const int wave = tid >> 5;
  const int lane = tid & 31;
  const int lh   = lane >> 4;
  const int l16  = lane & 15;
  const int qt = blockIdx.x * 8 + wave;   // 16-row query tile index
  const int bh = blockIdx.y;
  const int b  = bh / NQ_H;
  const int h  = bh % NQ_H;
  const int kv = h % NKV_H;               // faithful group-major GQA mapping
  const float scale = 0.125f;             // 1/sqrt(64)

  // Q as two A-fragments (K-steps over HD=64)
  v16bf aq[2];
  {
    const int s = qt * 16 + l16;
    const bf16* qrow = q + (((size_t)(b * SEQ + s) * NQ_H + h) * HEAD_D);
#pragma unroll
    for (int ks = 0; ks < 2; ++ks) {
      const bf16* p = qrow + ks * 32 + lh * 8;
      v8bf lo = *(const v8bf*)p;
      v8bf hi = *(const v8bf*)(p + 16);
      aq[ks] = pack16(lo, hi);
    }
  }

  float mrow[8], lrow[8];
  v8f o[4];
#pragma unroll
  for (int r = 0; r < 8; ++r) { mrow[r] = -1e30f; lrow[r] = 0.f; }
#pragma unroll
  for (int t = 0; t < 4; ++t) o[t] = zero8();

  const int nchunks = qt / 2 + 1;          // keys up to qt*16+15
  for (int jc = 0; jc < nchunks; ++jc) {
    // ---- scores: two 16x16 f32 tiles over a 32-key chunk ----
    v8f sc[2];
#pragma unroll
    for (int nt = 0; nt < 2; ++nt) {
      sc[nt] = zero8();
      const int key = jc * 32 + nt * 16 + l16;
      const bf16* krow = k + (((size_t)(b * SEQ + key) * NKV_H + kv) * HEAD_D);
#pragma unroll
      for (int ks = 0; ks < 2; ++ks) {
        const bf16* p = krow + ks * 32 + lh * 8;
        v8bf lo = *(const v8bf*)p;
        v8bf hi = *(const v8bf*)(p + 16);
        v16bf bk = pack16(lo, hi);
        sc[nt] = __builtin_amdgcn_wmma_f32_16x16x32_bf16(
            false, aq[ks], false, bk, (short)0, sc[nt], false, false);
      }
    }
    // ---- scale + causal mask + online softmax ----
    const int qi = qt * 16 + 8 * lh;       // + r below is the global query row
    float lm[8];
#pragma unroll
    for (int r = 0; r < 8; ++r) {
      float s0 = sc[0][r] * scale;
      float s1 = sc[1][r] * scale;
      if (jc * 32 + l16 > qi + r)      s0 = -1e30f;
      if (jc * 32 + 16 + l16 > qi + r) s1 = -1e30f;
      sc[0][r] = s0; sc[1][r] = s1;
      lm[r] = fmaxf(s0, s1);
    }
#pragma unroll
    for (int off = 8; off > 0; off >>= 1)
#pragma unroll
      for (int r = 0; r < 8; ++r)
        lm[r] = fmaxf(lm[r], __shfl_xor(lm[r], off, 16));
    float al[8], rs[8];
#pragma unroll
    for (int r = 0; r < 8; ++r) {
      float mn = fmaxf(mrow[r], lm[r]);
      al[r] = __expf(mrow[r] - mn);
      mrow[r] = mn;
      float p0 = __expf(sc[0][r] - mn);
      float p1 = __expf(sc[1][r] - mn);
      sc[0][r] = p0; sc[1][r] = p1;
      rs[r] = p0 + p1;
    }
#pragma unroll
    for (int off = 8; off > 0; off >>= 1)
#pragma unroll
      for (int r = 0; r < 8; ++r)
        rs[r] += __shfl_xor(rs[r], off, 16);
#pragma unroll
    for (int r = 0; r < 8; ++r) lrow[r] = lrow[r] * al[r] + rs[r];
#pragma unroll
    for (int t = 0; t < 4; ++t)
#pragma unroll
      for (int r = 0; r < 8; ++r) o[t][r] *= al[r];

    // ---- redistribute P: C layout -> A layout through per-wave LDS ----
#pragma unroll
    for (int nt = 0; nt < 2; ++nt)
#pragma unroll
      for (int r = 0; r < 8; ++r)
        Plds[wave][r + 8 * lh][nt * 16 + l16] = (bf16)sc[nt][r];
    asm volatile("s_wait_dscnt 0x0" ::: "memory");  // per-wave LDS fence
    v16bf ap;
    {
      const bf16* p = &Plds[wave][l16][lh * 8];
      v8bf lo = *(const v8bf*)p;
      v8bf hi = *(const v8bf*)(p + 16);
      ap = pack16(lo, hi);
    }
    // ---- O += P @ V (single K=32 WMMA per 16-wide d-tile) ----
#pragma unroll
    for (int nt = 0; nt < 4; ++nt) {
      const int d = nt * 16 + l16;
      const bf16* p = vT + (((size_t)(b * NKV_H + kv) * HEAD_D + d) * SEQ)
                       + jc * 32 + lh * 8;
      v8bf lo = *(const v8bf*)p;
      v8bf hi = *(const v8bf*)(p + 16);
      v16bf bv = pack16(lo, hi);
      o[nt] = __builtin_amdgcn_wmma_f32_16x16x32_bf16(
          false, ap, false, bv, (short)0, o[nt], false, false);
    }
  }
  // ---- normalize + store fp32 [B,S,NQ,HD] ----
  float rl[8];
#pragma unroll
  for (int r = 0; r < 8; ++r) rl[r] = 1.f / lrow[r];
#pragma unroll
  for (int nt = 0; nt < 4; ++nt)
#pragma unroll
    for (int r = 0; r < 8; ++r) {
      const int s = qt * 16 + r + 8 * lh;
      const int d = nt * 16 + l16;
      out[((size_t)(b * SEQ + s) * NQ_H + h) * HEAD_D + d] = o[nt][r] * rl[r];
    }
}

// ---------------------------------------------------------------------------
// Global-L2 norm helpers, RoPE+pack, V transpose pack, SwiGLU
// ---------------------------------------------------------------------------
__global__ void sumsq_kernel(const float* __restrict__ x, float* __restrict__ out, int n) {
  float s = 0.f;
  for (int i = blockIdx.x * blockDim.x + threadIdx.x; i < n; i += gridDim.x * blockDim.x) {
    float v = x[i];
    s += v * v;
  }
#pragma unroll
  for (int off = 16; off > 0; off >>= 1) s += __shfl_xor(s, off, 32);
  __shared__ float red[8];
  if ((threadIdx.x & 31) == 0) red[threadIdx.x >> 5] = s;
  __syncthreads();
  if (threadIdx.x == 0) {
    float t = 0.f;
    for (int i = 0; i < 8; ++i) t += red[i];
    atomicAdd(out, t);
  }
}

__global__ void gnorm_apply_kernel(const float* __restrict__ x, const float* __restrict__ w,
                                   const float* __restrict__ ss, float* __restrict__ y, int n) {
  int i = blockIdx.x * blockDim.x + threadIdx.x;
  if (i < n) {
    float rn = 1.f / (sqrtf(*ss) + 1e-6f);
    y[i] = w[i & (D_MODEL - 1)] * x[i] * rn;
  }
}

// interleaved RoPE on fp32 [rows, nh, 64], packed to bf16 (start_pos == 0)
__global__ void rope_pack_kernel(const float* __restrict__ in, bf16* __restrict__ out,
                                 const float* __restrict__ cs, const float* __restrict__ sn,
                                 int nh) {
  int idx = blockIdx.x * blockDim.x + threadIdx.x;
  int total = M_ROWS * nh * (HEAD_D / 2);
  if (idx >= total) return;
  int i   = idx & 31;            // HD/2 = 32
  int t   = idx >> 5;
  int hh  = t % nh;
  int row = t / nh;
  int s   = row & (SEQ - 1);
  float c  = cs[s * 32 + i];
  float si = sn[s * 32 + i];
  size_t base = ((size_t)row * nh + hh) * HEAD_D + 2 * i;
  float t0 = in[base], t1 = in[base + 1];
  out[base]     = (bf16)(t0 * c - t1 * si);
  out[base + 1] = (bf16)(t0 * si + t1 * c);
}

// fp32 [B,S,NKV,HD] -> bf16 [B,NKV,HD,S] (so V B-fragments are contiguous)
__global__ void pack_vT_kernel(const float* __restrict__ v, bf16* __restrict__ vT) {
  int idx = blockIdx.x * blockDim.x + threadIdx.x;
  if (idx >= M_ROWS * NKV_H * HEAD_D) return;
  int d   = idx & 63;
  int kvh = (idx >> 6) & 3;
  int row = idx >> 8;
  int b   = row >> 10;           // / SEQ
  int s   = row & (SEQ - 1);
  vT[(((size_t)b * NKV_H + kvh) * HEAD_D + d) * SEQ + s] = (bf16)v[idx];
}

__global__ void silu_mul_kernel(float* __restrict__ g0, const float* __restrict__ g1, int n) {
  int i = blockIdx.x * blockDim.x + threadIdx.x;
  if (i < n) {
    float x = g1[i];
    g0[i] = g0[i] * (x / (1.f + __expf(-x)));
  }
}

// ---------------------------------------------------------------------------
// Host orchestration (graph-capture safe: only async stream ops)
// ---------------------------------------------------------------------------
extern "C" void kernel_launch(void* const* d_in, const int* in_sizes, int n_in,
                              void* d_out, int out_size, void* d_ws, size_t ws_size,
                              hipStream_t stream) {
  (void)in_sizes; (void)n_in; (void)out_size; (void)ws_size;
  const float* x          = (const float*)d_in[0];
  const float* att_norm_w = (const float*)d_in[1];
  const float* qw    = (const float*)d_in[2];
  const float* qbias = (const float*)d_in[3];
  const float* kw    = (const float*)d_in[4];
  const float* kbias = (const float*)d_in[5];
  const float* vw    = (const float*)d_in[6];
  const float* vbias = (const float*)d_in[7];
  const float* ow    = (const float*)d_in[8];
  const float* obias = (const float*)d_in[9];
  const float* ffn_norm_w = (const float*)d_in[10];
  const float* w0 = (const float*)d_in[11];
  const float* b0 = (const float*)d_in[12];
  const float* w1 = (const float*)d_in[13];
  const float* b1 = (const float*)d_in[14];
  const float* w2 = (const float*)d_in[15];
  const float* b2 = (const float*)d_in[16];
  const float* out_norm_w = (const float*)d_in[17];
  const float* fcos = (const float*)d_in[18];
  const float* fsin = (const float*)d_in[19];
  // d_in[20] = start_pos (0 for this problem)

  char* wsb = (char*)d_ws;
  const size_t MB = 1024ull * 1024ull;
  float* h    = (float*)(wsb + 0);         // [2048,1024] residual stream
  float* xn   = (float*)(wsb + 8  * MB);   // normed activations
  float* qf   = (float*)(wsb + 16 * MB);   // q fp32
  float* kf   = (float*)(wsb + 24 * MB);   // k fp32
  float* vf   = (float*)(wsb + 26 * MB);   // v fp32
  float* attn = (float*)(wsb + 28 * MB);   // attention output fp32
  float* g0   = (float*)(wsb + 36 * MB);   // ffn gate
  float* g1   = (float*)(wsb + 68 * MB);   // ffn up
  bf16*  qb16 = (bf16*)(wsb + 100 * MB);
  bf16*  kb16 = (bf16*)(wsb + 104 * MB);
  bf16*  vT16 = (bf16*)(wsb + 105 * MB);
  float* ssp  = (float*)(wsb + 106 * MB);  // sum-of-squares scalar

  const int MD = M_ROWS * D_MODEL;
  hipMemcpyAsync(h, x, (size_t)MD * sizeof(float), hipMemcpyDeviceToDevice, stream);

  dim3 blk(256);
  for (int l = 0; l < N_LAYER; ++l) {
    // attention gnorm (global L2 over whole tensor)
    hipMemsetAsync(ssp, 0, sizeof(float), stream);
    sumsq_kernel<<<512, blk, 0, stream>>>(h, ssp, MD);
    gnorm_apply_kernel<<<(MD + 255) / 256, blk, 0, stream>>>(
        h, att_norm_w + (size_t)l * D_MODEL, ssp, xn, MD);
    // QKV projections (WMMA)
    gemm_bf16<<<dim3(8, 16), blk, 0, stream>>>(
        xn, qw + (size_t)l * D_MODEL * D_MODEL, qbias + (size_t)l * D_MODEL,
        nullptr, qf, M_ROWS, D_MODEL, D_MODEL);
    gemm_bf16<<<dim3(2, 16), blk, 0, stream>>>(
        xn, kw + (size_t)l * D_MODEL * 256, kbias + (size_t)l * 256,
        nullptr, kf, M_ROWS, 256, D_MODEL);
    gemm_bf16<<<dim3(2, 16), blk, 0, stream>>>(
        xn, vw + (size_t)l * D_MODEL * 256, vbias + (size_t)l * 256,
        nullptr, vf, M_ROWS, 256, D_MODEL);
    // RoPE + bf16 packing
    rope_pack_kernel<<<(M_ROWS * NQ_H * 32 + 255) / 256, blk, 0, stream>>>(
        qf, qb16, fcos, fsin, NQ_H);
    rope_pack_kernel<<<(M_ROWS * NKV_H * 32 + 255) / 256, blk, 0, stream>>>(
        kf, kb16, fcos, fsin, NKV_H);
    pack_vT_kernel<<<(M_ROWS * NKV_H * HEAD_D + 255) / 256, blk, 0, stream>>>(vf, vT16);
    // flash attention (WMMA)
    attn_kernel<<<dim3(8, BATCH * NQ_H), blk, 0, stream>>>(qb16, kb16, vT16, attn);
    // output projection + residual
    gemm_bf16<<<dim3(8, 16), blk, 0, stream>>>(
        attn, ow + (size_t)l * D_MODEL * D_MODEL, obias + (size_t)l * D_MODEL,
        h, h, M_ROWS, D_MODEL, D_MODEL);
    // ffn gnorm
    hipMemsetAsync(ssp, 0, sizeof(float), stream);
    sumsq_kernel<<<512, blk, 0, stream>>>(h, ssp, MD);
    gnorm_apply_kernel<<<(MD + 255) / 256, blk, 0, stream>>>(
        h, ffn_norm_w + (size_t)l * D_MODEL, ssp, xn, MD);
    // SwiGLU FFN
    gemm_bf16<<<dim3(32, 16), blk, 0, stream>>>(
        xn, w0 + (size_t)l * D_MODEL * FFN_H, b0 + (size_t)l * FFN_H,
        nullptr, g0, M_ROWS, FFN_H, D_MODEL);
    gemm_bf16<<<dim3(32, 16), blk, 0, stream>>>(
        xn, w1 + (size_t)l * D_MODEL * FFN_H, b1 + (size_t)l * FFN_H,
        nullptr, g1, M_ROWS, FFN_H, D_MODEL);
    silu_mul_kernel<<<(M_ROWS * FFN_H + 255) / 256, blk, 0, stream>>>(
        g0, g1, M_ROWS * FFN_H);
    gemm_bf16<<<dim3(8, 16), blk, 0, stream>>>(
        g0, w2 + (size_t)l * FFN_H * D_MODEL, b2 + (size_t)l * D_MODEL,
        h, h, M_ROWS, D_MODEL, FFN_H);
  }
  // final output gnorm -> d_out (fp32)
  hipMemsetAsync(ssp, 0, sizeof(float), stream);
  sumsq_kernel<<<512, blk, 0, stream>>>(h, ssp, MD);
  gnorm_apply_kernel<<<(MD + 255) / 256, blk, 0, stream>>>(
      h, out_norm_w, ssp, (float*)d_out, MD);
}
